// ScaledDotProductAttentionForT5_79517024518692
// MI455X (gfx1250) — compile-verified, matrix operand order checked by reading, not run on hardware
//
#include <hip/hip_runtime.h>
#include <hip/hip_bf16.h>
#include <stdint.h>

typedef unsigned int       u32;
typedef unsigned short     u16;
typedef unsigned long long u64;

typedef float  v8f   __attribute__((ext_vector_type(8)));
typedef u32    v8u   __attribute__((ext_vector_type(8)));
typedef u32    v4u   __attribute__((ext_vector_type(4)));
typedef __bf16 v16bf __attribute__((ext_vector_type(16)));
typedef u32    uv4   __attribute__((ext_vector_type(4)));
typedef int    iv4   __attribute__((ext_vector_type(4)));
typedef int    iv8   __attribute__((ext_vector_type(8)));

// address-space-qualified int4 pointers for the async-LDS builtins
typedef __attribute__((address_space(1))) iv4 as1_iv4;
typedef __attribute__((address_space(3))) iv4 as3_iv4;

#define T_LEN  2048
#define NHEAD  16
#define DKV    64
#define DMODEL 1024
#define INNER  1024
#define BATCH  2

// ---- CDNA5 data-mover feature detection (fall back to sync path if absent) ----
#if defined(__has_builtin)
#if __has_builtin(__builtin_amdgcn_global_load_async_to_lds_b128)
#define HAVE_ASYNC_LDS 1
#endif
#if __has_builtin(__builtin_amdgcn_tensor_load_to_lds)
#define HAVE_TDM 1
#endif
#endif

// ---------------- helpers ----------------

__device__ __forceinline__ u16 f2bf(float f) {
  u32 u = __builtin_bit_cast(u32, f);
  u32 r = 0x7FFFu + ((u >> 16) & 1u);
  return (u16)((u + r) >> 16);
}

// 16-byte copy global->LDS: async (ASYNCcnt) when available, else through VGPRs.
__device__ __forceinline__ void cp16_g2l(u16* lds, const u16* g) {
#ifdef HAVE_ASYNC_LDS
  __builtin_amdgcn_global_load_async_to_lds_b128(
      (as1_iv4*)(g), (as3_iv4*)(lds), 0, 0);
#else
  *(v4u*)lds = *(const v4u*)g;
#endif
}
__device__ __forceinline__ void cp_g2l_fence() {
#ifdef HAVE_ASYNC_LDS
  asm volatile("s_wait_asynccnt 0x0" ::: "memory");
#endif
}

#ifdef HAVE_TDM
// TDM 2-D tile load: descriptor per CDNA5 ISA §8.3/§8.4. count=1, type=2('image'),
// data_size code 1 (2 bytes). g2/g3 (3D..5D dims) zeroed / disabled.
__device__ __forceinline__ void tdm_load2d(uv4 g0, iv8 g1) {
  iv4 z4 = {0, 0, 0, 0};
#if __clang_major__ >= 23
  iv8 z8 = {0, 0, 0, 0, 0, 0, 0, 0};
  __builtin_amdgcn_tensor_load_to_lds(g0, g1, z4, z4, z8, 0);
#else
  __builtin_amdgcn_tensor_load_to_lds(g0, g1, z4, z4, 0);
#endif
}
__device__ __forceinline__ uv4 tdm_g0(u32 lds_addr, u64 gaddr) {
  uv4 g0 = { 1u,                      // count=1 (valid), is_restore=0, gather=0
             lds_addr,                // LDS byte address
             (u32)gaddr,              // global_addr[31:0]
             ((u32)(gaddr >> 32) & 0x01FFFFFFu) | 0x80000000u }; // addr[56:32], type=2
  return g0;
}
// dim0 = inner (contiguous) tensor length, dim1 = outer length,
// stride0 = elements between rows, tile0 x tile1 = tile shape (elements).
__device__ __forceinline__ iv8 tdm_g1(u32 dim0, u32 dim1, u32 stride0,
                                      u32 tile0, u32 tile1) {
  iv8 g1 = { (int)0x00010000u,             // workgroup_mask=0, data_size=1 (2B)
             (int)(dim0 << 16),            // tensor_dim0[15:0]
             (int)((dim0 >> 16) | (dim1 << 16)),
             (int)((dim1 >> 16) | (tile0 << 16)),
             (int)(tile1 & 0xFFFFu),       // tile_dim1 ([143:128]); tile_dim2=0
             (int)stride0,                 // tensor_dim0_stride[31:0]
             (int)((stride0 >> 16) >> 16), // stride0[47:32] | dim1_stride lo (0)
             0 };
  return g1;
}
#endif

// Load one 16x32 bf16 A/B fragment (CDNA5 §7.12.2 16-bit layout):
// lane L holds row (L%16); K-chunks at kb=(L<16?0:8) and kb+16, 8 elems each.
__device__ __forceinline__ v16bf frag_ld(const u16* base, int row, int rowStride) {
  int lane = threadIdx.x & 31;
  int kb = (lane < 16) ? 0 : 8;
  const u16* p = base + (size_t)row * rowStride + kb;
  v4u a = *(const v4u*)p;
  v4u b = *(const v4u*)(p + 16);
  v8u u = { a.x, a.y, a.z, a.w, b.x, b.y, b.z, b.w };
  return __builtin_bit_cast(v16bf, u);
}

// 16-lane reductions via ds_swizzle immediates (xor 1/2/4/8 stay inside each
// 16-lane half of the WMMA C layout). offset = xor<<10 | and_mask 0x1f.
#define SWZ(x, imm) __builtin_bit_cast(float, __builtin_amdgcn_ds_swizzle(__builtin_bit_cast(int, (x)), (imm)))
__device__ __forceinline__ float red16_max(float x) {
  x = fmaxf(x, SWZ(x, 0x041F));
  x = fmaxf(x, SWZ(x, 0x081F));
  x = fmaxf(x, SWZ(x, 0x101F));
  x = fmaxf(x, SWZ(x, 0x201F));
  return x;
}
__device__ __forceinline__ float red16_sum(float x) {
  x += SWZ(x, 0x041F);
  x += SWZ(x, 0x081F);
  x += SWZ(x, 0x101F);
  x += SWZ(x, 0x201F);
  return x;
}

// T5 relative position bias (bidirectional, 32 buckets, max_distance=128)
__device__ __forceinline__ float t5_bias_val(const float* rel, int h, int t, int k) {
  int d = k - t;
  int ret = (d > 0) ? 16 : 0;
  int ad = (d < 0) ? -d : d;
  int bucket;
  if (ad < 8) {
    bucket = ad;
  } else {
    // 8 + int(log(ad/8) * 8/log(16)); 8/ln(16) = 2.8853900817779268
    int rl = 8 + (int)(__logf((float)ad * 0.125f) * 2.8853900817779268f);
    bucket = (rl < 15) ? rl : 15;
  }
  return rel[(ret + bucket) * NHEAD + h];
}

// ---------------- conversion kernels ----------------

__global__ __launch_bounds__(256)
void f32_to_bf16_vec(const float* __restrict__ in, u16* __restrict__ out, int n) {
  int i = blockIdx.x * 256 + threadIdx.x;
  if (i < n) out[i] = f2bf(in[i]);
}

// W [1024][1024] fp32 (K-major)  ->  Wt [N][K] bf16 (transposed)
__global__ __launch_bounds__(256)
void transpose_f32_to_bf16(const float* __restrict__ W, u16* __restrict__ Wt) {
  __shared__ float tile[32][33];
  int bx = blockIdx.x * 32;           // n block
  int by = blockIdx.y * 32;           // k block
  int tx = threadIdx.x & 31;
  int ty = threadIdx.x >> 5;          // 0..7
  #pragma unroll
  for (int i = 0; i < 4; ++i)
    tile[ty + i * 8][tx] = W[(size_t)(by + ty + i * 8) * 1024 + bx + tx];
  __syncthreads();
  #pragma unroll
  for (int i = 0; i < 4; ++i)
    Wt[(size_t)(bx + ty + i * 8) * 1024 + by + tx] = f2bf(tile[tx][ty + i * 8]);
}

// ---------------- bf16 WMMA GEMM ----------------
// C[M=4096][N=1024] = A[M][K=1024] * Bt[N][K]^T
// MODE 0: fp32 row-major out;  MODE 1: bf16 [B][H][T][64];  MODE 2: bf16 [B][H][64][T]
template<int MODE>
__global__ __launch_bounds__(256, 2)
void gemm_bf16_wmma(const u16* __restrict__ A, const u16* __restrict__ Bt,
                    void* __restrict__ out) {
  constexpr int Kdim = 1024, Ndim = 1024;
  __shared__ __attribute__((aligned(16))) u16 As[128 * 32];
  __shared__ __attribute__((aligned(16))) u16 Bs[128 * 32];

  const int tid  = threadIdx.x;
  const int lane = tid & 31;
  const int w    = tid >> 5;
  const int wm   = w >> 1;            // 0..3  (M waves)
  const int wn   = w & 1;             // 0..1  (N waves)
  const int m0   = blockIdx.y * 128;
  const int n0   = blockIdx.x * 128;
  const int ln   = lane & 15;

  const v8f z8 = {0.f,0.f,0.f,0.f,0.f,0.f,0.f,0.f};
  v8f acc[2][4];
  #pragma unroll
  for (int i = 0; i < 2; ++i)
    #pragma unroll
    for (int j = 0; j < 4; ++j) acc[i][j] = z8;

  for (int k0 = 0; k0 < Kdim; k0 += 32) {
    #pragma unroll
    for (int p = 0; p < 2; ++p) {
      int row = p * 64 + (tid >> 2);
      int c8  = (tid & 3) * 8;
      cp16_g2l(As + row * 32 + c8, A  + (size_t)(m0 + row) * Kdim + k0 + c8);
      cp16_g2l(Bs + row * 32 + c8, Bt + (size_t)(n0 + row) * Kdim + k0 + c8);
    }
    cp_g2l_fence();
    __syncthreads();

    v16bf af[2], bfr[4];
    #pragma unroll
    for (int mt = 0; mt < 2; ++mt) af[mt]  = frag_ld(As, wm * 32 + mt * 16 + ln, 32);
    #pragma unroll
    for (int nt = 0; nt < 4; ++nt) bfr[nt] = frag_ld(Bs, wn * 64 + nt * 16 + ln, 32);
    #pragma unroll
    for (int mt = 0; mt < 2; ++mt)
      #pragma unroll
      for (int nt = 0; nt < 4; ++nt)
        acc[mt][nt] = __builtin_amdgcn_wmma_f32_16x16x32_bf16(
            false, af[mt], false, bfr[nt], (short)0, acc[mt][nt], false, false);
    __syncthreads();
  }

  const int hi8 = (lane >= 16) ? 8 : 0;
  #pragma unroll
  for (int mt = 0; mt < 2; ++mt) {
    #pragma unroll
    for (int nt = 0; nt < 4; ++nt) {
      #pragma unroll
      for (int r = 0; r < 8; ++r) {
        int m = m0 + wm * 32 + mt * 16 + r + hi8;
        int n = n0 + wn * 64 + nt * 16 + ln;
        float v = acc[mt][nt][r];
        if (MODE == 0) {
          ((float*)out)[(size_t)m * Ndim + n] = v;
        } else {
          int b = m >> 11, t = m & (T_LEN - 1);
          int h = n >> 6,  d = n & 63;
          size_t idx = (MODE == 1)
              ? ((((size_t)b * NHEAD + h) * T_LEN + t) * DKV + d)
              : ((((size_t)b * NHEAD + h) * DKV + d) * T_LEN + t);
          ((u16*)out)[idx] = f2bf(v);
        }
      }
    }
  }
}

// ---------------- flash attention (T5, unscaled, + relative bias) ----------------
// Q,K: [B][H][T][64] bf16;  Vt: [B][H][64][T] bf16;  Cbf out: [B][T][H*64] bf16
__global__ __launch_bounds__(256, 2)
void flash_attn_t5(const u16* __restrict__ Q, const u16* __restrict__ Kg,
                   const u16* __restrict__ Vt, const float* __restrict__ rel_emb,
                   u16* __restrict__ Cbf) {
  __shared__ __attribute__((aligned(16))) u16 Ks[32 * 64];       // [s=32][d=64]
  __shared__ __attribute__((aligned(16))) u16 Vs[64 * 32];       // [d=64][s=32]
  __shared__ __attribute__((aligned(16))) u16 Ps[8 * 16 * 32];   // per-wave P tile
  __shared__ float relS[32 * NHEAD];

  const int tid  = threadIdx.x;
  const int lane = tid & 31;
  const int w    = tid >> 5;
  const int ln   = lane & 15;
  const int hi8  = (lane >= 16) ? 8 : 0;
  const int b    = blockIdx.z;
  const int h    = blockIdx.y;
  const int q0   = blockIdx.x * 128 + w * 16;

  for (int i = tid; i < 32 * NHEAD; i += 256) relS[i] = rel_emb[i];

  const u16* Qb  = Q  + ((size_t)(b * NHEAD + h) * T_LEN) * DKV;
  const u16* Kb  = Kg + ((size_t)(b * NHEAD + h) * T_LEN) * DKV;
  const u16* Vtb = Vt + ((size_t)(b * NHEAD + h) * DKV) * T_LEN;

  // Q fragments for this wave's 16 rows (k = 0..31 and 32..63)
  v16bf qf[2];
  {
    int row = q0 + ln;
    int kb = (lane < 16) ? 0 : 8;
    #pragma unroll
    for (int kk = 0; kk < 2; ++kk) {
      const u16* p = Qb + (size_t)row * DKV + kk * 32 + kb;
      v4u a = *(const v4u*)p;
      v4u c = *(const v4u*)(p + 16);
      v8u u = { a.x, a.y, a.z, a.w, c.x, c.y, c.z, c.w };
      qf[kk] = __builtin_bit_cast(v16bf, u);
    }
  }

  const v8f z8 = {0.f,0.f,0.f,0.f,0.f,0.f,0.f,0.f};
  v8f oacc[4] = { z8, z8, z8, z8 };
  float m_i[8], l_i[8];
  #pragma unroll
  for (int r = 0; r < 8; ++r) { m_i[r] = -3.0e38f; l_i[r] = 0.f; }

  u16* PsW = Ps + w * 16 * 32;

  for (int s0 = 0; s0 < T_LEN; s0 += 32) {
    __syncthreads();
#ifdef HAVE_TDM
    // Tensor Data Mover: one 2-D descriptor each for the K (32x64) and V^T
    // (64x32) tiles. Issued by wave 0 only (TDM ignores EXEC, per-wave op).
    if (w == 0) {
      u64 gaK = (u64)(uintptr_t)Kb  + (u64)s0 * (DKV * 2);
      u64 gaV = (u64)(uintptr_t)Vtb + (u64)s0 * 2;
      tdm_load2d(tdm_g0((u32)(size_t)&Ks[0], gaK),
                 tdm_g1(/*dim0*/DKV,   /*dim1*/T_LEN, /*stride0*/DKV,
                        /*tile0*/DKV,  /*tile1*/32));
      tdm_load2d(tdm_g0((u32)(size_t)&Vs[0], gaV),
                 tdm_g1(/*dim0*/T_LEN, /*dim1*/DKV,   /*stride0*/T_LEN,
                        /*tile0*/32,   /*tile1*/DKV));
      __builtin_amdgcn_s_wait_tensorcnt(0);
    }
#else
    {   // cooperative stage of K (32x64) and V^T (64x32) tiles
      int row = tid >> 3, c8 = (tid & 7) * 8;
      cp16_g2l(Ks + row * 64 + c8, Kb + (size_t)(s0 + row) * DKV + c8);
      int vrow = tid >> 2, vc8 = (tid & 3) * 8;
      cp16_g2l(Vs + vrow * 32 + vc8, Vtb + (size_t)vrow * T_LEN + s0 + vc8);
      if (s0 + 32 < T_LEN) {
        __builtin_prefetch(Kb + (size_t)(s0 + 32 + row) * DKV + c8, 0, 3);
        __builtin_prefetch(Vtb + (size_t)vrow * T_LEN + s0 + 32 + vc8, 0, 3);
      }
      cp_g2l_fence();
    }
#endif
    __syncthreads();

    // S tiles: sA = Q·K^T cols [s0,s0+16), sB = cols [s0+16,s0+32)
    v8f sA = z8, sB = z8;
    #pragma unroll
    for (int kk = 0; kk < 2; ++kk) {
      v16bf kf0 = frag_ld(Ks + kk * 32, ln,      64);
      v16bf kf1 = frag_ld(Ks + kk * 32, 16 + ln, 64);
      sA = __builtin_amdgcn_wmma_f32_16x16x32_bf16(false, qf[kk], false, kf0, (short)0, sA, false, false);
      sB = __builtin_amdgcn_wmma_f32_16x16x32_bf16(false, qf[kk], false, kf1, (short)0, sB, false, false);
    }

    // bias + online softmax (row r lives at VGPR r, halves give rows r / r+8)
    #pragma unroll
    for (int r = 0; r < 8; ++r) {
      int t = q0 + r + hi8;
      float a = sA[r] + t5_bias_val(relS, h, t, s0 + ln);
      float c = sB[r] + t5_bias_val(relS, h, t, s0 + 16 + ln);
      float mx  = red16_max(fmaxf(a, c));
      float mnew = fmaxf(m_i[r], mx);
      float pA = __expf(a - mnew);
      float pB = __expf(c - mnew);
      float rs = red16_sum(pA + pB);
      float scale = __expf(m_i[r] - mnew);
      l_i[r] = l_i[r] * scale + rs;
      m_i[r] = mnew;
      #pragma unroll
      for (int dt = 0; dt < 4; ++dt) oacc[dt][r] *= scale;
      // store P row-major to per-wave LDS (C layout -> A layout shuffle)
      PsW[(r + hi8) * 32 + ln]      = f2bf(pA);
      PsW[(r + hi8) * 32 + 16 + ln] = f2bf(pB);
    }
    asm volatile("s_wait_dscnt 0x0" ::: "memory");

    v16bf pf = frag_ld(PsW, ln, 32);
    #pragma unroll
    for (int dt = 0; dt < 4; ++dt) {
      v16bf vf = frag_ld(Vs, dt * 16 + ln, 32);
      oacc[dt] = __builtin_amdgcn_wmma_f32_16x16x32_bf16(
          false, pf, false, vf, (short)0, oacc[dt], false, false);
    }
  }

  // epilogue: normalize and write context, layout [B][T][h*64+d]
  #pragma unroll
  for (int r = 0; r < 8; ++r) {
    float inv = 1.0f / l_i[r];
    int t = q0 + r + hi8;
    u16* dst = Cbf + ((size_t)(b * T_LEN + t)) * INNER + h * DKV;
    #pragma unroll
    for (int dt = 0; dt < 4; ++dt)
      dst[dt * 16 + ln] = f2bf(oacc[dt][r] * inv);
  }
}

// ---------------- position bias output (pure bandwidth) ----------------
__global__ __launch_bounds__(256)
void bias_out_kernel(const float* __restrict__ rel_emb, float* __restrict__ biasOut) {
  size_t idx = (size_t)blockIdx.x * 256 + threadIdx.x;   // [H][T][K] flat
  int k = (int)(idx & (T_LEN - 1));
  int t = (int)((idx >> 11) & (T_LEN - 1));
  int h = (int)(idx >> 22);
  biasOut[idx] = t5_bias_val(rel_emb, h, t, k);
}

// ---------------- launch ----------------
extern "C" void kernel_launch(void* const* d_in, const int* in_sizes, int n_in,
                              void* d_out, int out_size, void* d_ws, size_t ws_size,
                              hipStream_t stream) {
  (void)in_sizes; (void)n_in; (void)out_size; (void)ws_size;
  const float* x   = (const float*)d_in[0];
  const float* Wq  = (const float*)d_in[1];
  const float* Wk  = (const float*)d_in[2];
  const float* Wv  = (const float*)d_in[3];
  const float* Wo  = (const float*)d_in[4];
  const float* rel = (const float*)d_in[5];

  char* ws = (char*)d_ws;
  size_t off = 0;
  auto alloc = [&](size_t bytes) -> char* {
    char* p = ws + off;
    off = (off + bytes + 255) & ~(size_t)255;
    return p;
  };

  u16* Xbf = (u16*)alloc((size_t)BATCH * T_LEN * DMODEL * 2);
  u16* WqT = (u16*)alloc((size_t)DMODEL * INNER * 2);
  u16* WkT = (u16*)alloc((size_t)DMODEL * INNER * 2);
  u16* WvT = (u16*)alloc((size_t)DMODEL * INNER * 2);
  u16* WoT = (u16*)alloc((size_t)INNER * DMODEL * 2);
  u16* Qbf = (u16*)alloc((size_t)BATCH * NHEAD * T_LEN * DKV * 2);
  u16* Kbf = (u16*)alloc((size_t)BATCH * NHEAD * T_LEN * DKV * 2);
  u16* Vtb = (u16*)alloc((size_t)BATCH * NHEAD * DKV * T_LEN * 2);
  u16* Cbf = (u16*)alloc((size_t)BATCH * T_LEN * INNER * 2);

  float* outO    = (float*)d_out;                                  // [B][T][D_MODEL]
  float* outBias = outO + (size_t)BATCH * T_LEN * DMODEL;          // [1][H][T][K]

  int nX = BATCH * T_LEN * DMODEL;
  f32_to_bf16_vec<<<(nX + 255) / 256, 256, 0, stream>>>(x, Xbf, nX);

  dim3 tg(32, 32);
  transpose_f32_to_bf16<<<tg, 256, 0, stream>>>(Wq, WqT);
  transpose_f32_to_bf16<<<tg, 256, 0, stream>>>(Wk, WkT);
  transpose_f32_to_bf16<<<tg, 256, 0, stream>>>(Wv, WvT);
  transpose_f32_to_bf16<<<tg, 256, 0, stream>>>(Wo, WoT);

  dim3 gg(INNER / 128, (BATCH * T_LEN) / 128);                     // (8, 32)
  gemm_bf16_wmma<1><<<gg, 256, 0, stream>>>(Xbf, WqT, Qbf);
  gemm_bf16_wmma<1><<<gg, 256, 0, stream>>>(Xbf, WkT, Kbf);
  gemm_bf16_wmma<2><<<gg, 256, 0, stream>>>(Xbf, WvT, Vtb);

  dim3 fg(T_LEN / 128, NHEAD, BATCH);                              // (16, 16, 2)
  flash_attn_t5<<<fg, 256, 0, stream>>>(Qbf, Kbf, Vtb, rel, Cbf);

  dim3 go(DMODEL / 128, (BATCH * T_LEN) / 128);
  gemm_bf16_wmma<0><<<go, 256, 0, stream>>>(Cbf, WoT, outO);

  size_t nBias = (size_t)NHEAD * T_LEN * T_LEN;
  bias_out_kernel<<<(unsigned)(nBias / 256), 256, 0, stream>>>(rel, outBias);
}
